// Quantisation_39848706572551
// MI455X (gfx1250) — matrix operand order repaired and checked by reading, not compiled, not run on hardware
//
#include <hip/hip_runtime.h>

typedef __attribute__((ext_vector_type(16))) __bf16       v16bf;
typedef __attribute__((ext_vector_type(8)))  float        v8f;
typedef __attribute__((ext_vector_type(8)))  unsigned int v8u;

#define DIM        256
#define NCODES     1024
#define ROWS_PB    128     // rows per block (8 waves x 16)
#define WTILE      32      // codes per LDS tile (= 2 n-tiles of 16)
#define NWT        (NCODES / WTILE)   // 32 W tiles

__device__ __forceinline__ void f32_to_bf16_pair(float f, __bf16& h, __bf16& l) {
    h = (__bf16)f;                 // round-to-nearest bf16 (high part)
    l = (__bf16)(f - (float)h);    // residual (low part)
}

__device__ __forceinline__ unsigned int pack_bf16(__bf16 lo, __bf16 hi) {
    unsigned int a = (unsigned int)__builtin_bit_cast(unsigned short, lo);
    unsigned int b = (unsigned int)__builtin_bit_cast(unsigned short, hi);
    return a | (b << 16);
}

__global__ __launch_bounds__(256) void vq_argmin_gather_kernel(
    const float* __restrict__ x,     // [N, 256]
    const float* __restrict__ w,     // [1024, 256]
    float* __restrict__ out)         // [N, 256]
{
    // B fragments for current W tile, pre-swizzled for v_wmma_f32_16x16x32_bf16:
    // [ntile][kstep][lane][vgpr j] ; lane&15 = column (code), lane>=16 -> K+8, j packs K=2j,2j+1 (+8 for j>=4)
    __shared__ __align__(32) unsigned int sBhi[2][8][32][8];   // 16 KB
    __shared__ __align__(32) unsigned int sBlo[2][8][32][8];   // 16 KB
    __shared__ float sWsq[NCODES];                             // 4 KB
    __shared__ int   sIdx[ROWS_PB];                            // 0.5 KB

    const int tid  = threadIdx.x;
    const int lane = tid & 31;
    const int wave = tid >> 5;
    const int l15  = lane & 15;
    const int hi16 = lane >> 4;            // 0 or 1
    const int row0 = blockIdx.x * ROWS_PB + wave * 16;

    // ---- 1. ||w_c||^2 for all 1024 codes (one pass, tiny vs main loop) ----
    for (int i = 0; i < 4; ++i) {
        int c = tid + i * 256;
        const float4* wr = reinterpret_cast<const float4*>(w + c * DIM);
        float s = 0.f;
        for (int q = 0; q < 64; ++q) {
            float4 v = wr[q];
            s += v.x * v.x + v.y * v.y + v.z * v.z + v.w * v.w;
        }
        sWsq[c] = s;
    }

    // ---- 2. Load this wave's 16x256 A-tile into registers as hi/lo bf16 fragments ----
    // 16-bit A 16x32 layout: lanes 0-15 (row = lane): VGPR0-3 K=0..7, VGPR4-7 K=16..23;
    // lanes 16-31 (row = lane-16): K=8..15 and K=24..31.
    v16bf Ahi[8], Alo[8];
    {
        const float* xr = x + (size_t)(row0 + l15) * DIM;
        const int off = hi16 * 8;
        for (int s = 0; s < 8; ++s) {
            const float4* p0 = reinterpret_cast<const float4*>(xr + s * 32 + off);
            const float4* p1 = reinterpret_cast<const float4*>(xr + s * 32 + off + 16);
            float c0[8], c1[8];
            float4 a0 = p0[0], a1 = p0[1], b0 = p1[0], b1 = p1[1];
            c0[0]=a0.x; c0[1]=a0.y; c0[2]=a0.z; c0[3]=a0.w;
            c0[4]=a1.x; c0[5]=a1.y; c0[6]=a1.z; c0[7]=a1.w;
            c1[0]=b0.x; c1[1]=b0.y; c1[2]=b0.z; c1[3]=b0.w;
            c1[4]=b1.x; c1[5]=b1.y; c1[6]=b1.z; c1[7]=b1.w;
            for (int e = 0; e < 8; ++e) {
                __bf16 h, l;
                f32_to_bf16_pair(c0[e], h, l);
                Ahi[s][e] = h; Alo[s][e] = l;
                f32_to_bf16_pair(c1[e], h, l);
                Ahi[s][8 + e] = h; Alo[s][8 + e] = l;
            }
        }
    }

    // ---- 3. Stream codebook through LDS; running argmin of ||w||^2 - 2 x.w ----
    float best[8];
    int   bidx[8];
    for (int j = 0; j < 8; ++j) { best[j] = 3.402823466e+38f; bidx[j] = 0; }

    for (int t = 0; t < NWT; ++t) {
        const int k0 = t * WTILE;
        __syncthreads();
        // Cooperative fill: 4096 hi dwords (+ matching lo), 16 per thread.
        for (int i = 0; i < 16; ++i) {
            int f  = tid * 16 + i;              // [0, 4096)
            int j  = f & 7;
            int bl = (f >> 3) & 31;
            int s  = (f >> 8) & 7;
            int nt = (f >> 11) & 1;
            int c   = k0 + nt * 16 + (bl & 15);
            int col = s * 32 + ((bl >> 4) ? 8 : 0) + 2 * j + ((j >= 4) ? 8 : 0);
            float w0 = w[(size_t)c * DIM + col];
            float w1 = w[(size_t)c * DIM + col + 1];
            __bf16 h0, l0, h1, l1;
            f32_to_bf16_pair(w0, h0, l0);
            f32_to_bf16_pair(w1, h1, l1);
            sBhi[nt][s][bl][j] = pack_bf16(h0, h1);
            sBlo[nt][s][bl][j] = pack_bf16(l0, l1);
        }
        __syncthreads();

        // Prefetch next W tile into cache while we compute on this one.
        if (t + 1 < NWT)
            __builtin_prefetch(w + (size_t)(k0 + WTILE) * DIM + tid * 32, 0, 1);

        for (int nt = 0; nt < 2; ++nt) {
            v8f acc = {};
            for (int s = 0; s < 8; ++s) {
                v16bf Bh = __builtin_bit_cast(v16bf,
                    *reinterpret_cast<const v8u*>(&sBhi[nt][s][lane][0]));
                v16bf Bl = __builtin_bit_cast(v16bf,
                    *reinterpret_cast<const v8u*>(&sBlo[nt][s][lane][0]));
                acc = __builtin_amdgcn_wmma_f32_16x16x32_bf16(
                          false, Ahi[s], false, Bh, (short)0, acc, false, false);
                acc = __builtin_amdgcn_wmma_f32_16x16x32_bf16(
                          false, Ahi[s], false, Bl, (short)0, acc, false, false);
                acc = __builtin_amdgcn_wmma_f32_16x16x32_bf16(
                          false, Alo[s], false, Bh, (short)0, acc, false, false);
            }
            // C layout: lane&15 = column n; VGPR j = row j (+8 for lanes>=16)
            const int c = k0 + nt * 16 + l15;
            const float wsq = sWsq[c];
            for (int j = 0; j < 8; ++j) {
                float sc = wsq - 2.0f * acc[j];
                if (sc < best[j]) { best[j] = sc; bidx[j] = c; }
            }
        }
    }

    // ---- 4. Cross-lane argmin over the 16 columns held per lane group ----
    for (int m = 1; m < 16; m <<= 1) {
        for (int j = 0; j < 8; ++j) {
            float ob = __shfl_xor(best[j], m, 32);
            int   oi = __shfl_xor(bidx[j], m, 32);
            if (ob < best[j] || (ob == best[j] && oi < bidx[j])) {
                best[j] = ob; bidx[j] = oi;
            }
        }
    }
    if (l15 == 0) {
        for (int j = 0; j < 8; ++j)
            sIdx[wave * 16 + hi16 * 8 + j] = bidx[j];
    }
    __syncthreads();

    // ---- 5. Gather winning codebook rows to output (2 threads per row) ----
    {
        const int r    = tid >> 1;     // 0..127
        const int half = tid & 1;
        const int code = sIdx[r];
        const float4* src = reinterpret_cast<const float4*>(w + (size_t)code * DIM + half * 128);
        float4* dst = reinterpret_cast<float4*>(out + (size_t)(blockIdx.x * ROWS_PB + r) * DIM + half * 128);
        for (int q = 0; q < 32; ++q) dst[q] = src[q];
    }
}

extern "C" void kernel_launch(void* const* d_in, const int* in_sizes, int n_in,
                              void* d_out, int out_size, void* d_ws, size_t ws_size,
                              hipStream_t stream) {
    const float* x = (const float*)d_in[0];          // [65536, 256]
    const float* w = (const float*)d_in[1];          // [1024, 256]
    float* out = (float*)d_out;                      // [65536, 256]
    const int n_rows = in_sizes[0] / DIM;            // 65536
    dim3 grid(n_rows / ROWS_PB);                     // 512 blocks
    vq_argmin_gather_kernel<<<grid, 256, 0, stream>>>(x, w, out);
}
